// ProbSparseAttention_12695923327599
// MI455X (gfx1250) — matrix operand order repaired and checked by reading, not compiled, not run on hardware
//
#include <hip/hip_runtime.h>
#include <hip/hip_bf16.h>

typedef __attribute__((ext_vector_type(16))) _Float16 v16h;
typedef __attribute__((ext_vector_type(8)))  _Float16 v8h;
typedef __attribute__((ext_vector_type(8)))  float    v8f;

__device__ inline v8f wmma_f16(v16h a, v16h b, v8f c) {
  return __builtin_amdgcn_wmma_f32_16x16x32_f16(
      /*neg_a=*/false, a, /*neg_b=*/false, b,
      /*c_mod=*/(short)0, c, /*reuse_a=*/false, /*reuse_b=*/false);
}

__device__ inline v16h cat8(v8h lo, v8h hi) {
  v16h r;
#pragma unroll
  for (int i = 0; i < 8; ++i) { r[i] = lo[i]; r[i + 8] = hi[i]; }
  return r;
}

// ---- CDNA5 async global->LDS copy (ASYNCcnt path), per-lane 16B ----
__device__ inline void async_copy_b128(unsigned lds_off, const void* gptr) {
  asm volatile("global_load_async_to_lds_b128 %0, %1, off"
               :: "v"(lds_off), "v"((unsigned long long)(size_t)gptr)
               : "memory");
}
__device__ inline void wait_async0() {
  asm volatile("s_wait_asynccnt 0x0" ::: "memory");
}

// ---- DPP16 butterfly reductions over each 16-lane row (pure VALU, no DS) ----
template <int CTRL>
__device__ inline float dpp_f(float x) {
  return __int_as_float(__builtin_amdgcn_update_dpp(
      __float_as_int(x), __float_as_int(x), CTRL, 0xF, 0xF, true));
}
__device__ inline float red16_max(float x) {
  x = fmaxf(x, dpp_f<0xB1>(x));   // quad_perm(1,0,3,2)  : xor 1
  x = fmaxf(x, dpp_f<0x4E>(x));   // quad_perm(2,3,0,1)  : xor 2
  x = fmaxf(x, dpp_f<0x141>(x));  // row_half_mirror     : 8-group
  x = fmaxf(x, dpp_f<0x140>(x));  // row_mirror          : 16-group
  return x;
}
__device__ inline float red16_sum(float x) {
  x += dpp_f<0xB1>(x);
  x += dpp_f<0x4E>(x);
  x += dpp_f<0x141>(x);
  x += dpp_f<0x140>(x);
  return x;
}

// A fragment (16x32 f16): lane16 = row; halves[0..7]  = row[kc + hi*16 .. +8)
//                                       halves[8..15] = row[kc+hi*16+8 .. +8)
__device__ inline v16h a_frag(const float* rowp, int hi) {
  v16h a;
#pragma unroll
  for (int i = 0; i < 8; ++i) a[i]     = (_Float16)rowp[hi * 16 + i];
#pragma unroll
  for (int i = 0; i < 8; ++i) a[i + 8] = (_Float16)rowp[hi * 16 + 8 + i];
  return a;
}
__device__ inline v16h a_frag(const _Float16* rowp, int hi) {
  v8h lo = *(const v8h*)(rowp + hi * 16);
  v8h hv = *(const v8h*)(rowp + hi * 16 + 8);
  return cat8(lo, hv);
}

// B fragment (32x16 f16): lane16 = column; halves[j] = col[kc + hi*16 + j].
__device__ inline v16h b_frag_cvt(const float* p) {
  v16h b;
#pragma unroll
  for (int i = 0; i < 16; ++i) b[i] = (_Float16)p[i];
  return b;
}
__device__ inline v16h b_frag_lds(const _Float16* p) {
  return cat8(*(const v8h*)p, *(const v8h*)(p + 8));
}

// ---------------------------------------------------------------------------
// Projection GEMM: Y = X @ W^T + bias.  X: (4096 x 512), W: (512 x 512).
// MODE 0: f16 out [b,h,l,e]   (Q, K)
// MODE 1: f16 out [b,h,e,l]   (V, transposed for P.V B-fragments)
// MODE 2: f32 out [row, col]  (final output projection)
// Block = 128 threads (4 waves); each block computes a 64x64 tile.
// ---------------------------------------------------------------------------
template <int MODE, typename XT>
__global__ void proj_gemm(const XT* __restrict__ X, const float* __restrict__ W,
                          const float* __restrict__ bias, void* __restrict__ out) {
  constexpr int D = 512, L = 2048, H = 8, E = 64;
  const int lane   = threadIdx.x & 31;
  const int w      = threadIdx.x >> 5;
  const int lane16 = lane & 15;
  const int hi     = lane >> 4;

  const int arow  = blockIdx.x * 64 + w * 16 + lane16;
  const int nbase = blockIdx.y * 64;

  v8f acc[4] = {};
  const XT* xrow = X + (size_t)arow * D;

  for (int kc = 0; kc < D; kc += 32) {
    v16h a = a_frag(xrow + kc, hi);
#pragma unroll
    for (int nb = 0; nb < 4; ++nb) {
      v16h b = b_frag_cvt(W + (size_t)(nbase + nb * 16 + lane16) * D + kc + hi * 16);
      acc[nb] = wmma_f16(a, b, acc[nb]);
    }
  }

  // C layout: vgpr r, lanes 0-15 -> M = r, lanes 16-31 -> M = r + 8; N = lane16.
  const int rowb = blockIdx.x * 64 + w * 16 + hi * 8;
#pragma unroll
  for (int nb = 0; nb < 4; ++nb) {
    const int col = nbase + nb * 16 + lane16;
    const float bv = bias[col];
#pragma unroll
    for (int r = 0; r < 8; ++r) {
      const int row = rowb + r;
      const float val = acc[nb][r] + bv;
      if (MODE == 2) {
        ((float*)out)[(size_t)row * D + col] = val;
      } else {
        const int b_ = row / L, l_ = row % L;
        const int h_ = col / E, e_ = col % E;
        const _Float16 hv = (_Float16)val;
        if (MODE == 0)
          ((_Float16*)out)[(((size_t)(b_ * H + h_)) * L + l_) * E + e_] = hv;
        else
          ((_Float16*)out)[(((size_t)(b_ * H + h_)) * E + e_) * L + l_] = hv;
      }
    }
  }
}

// ---------------------------------------------------------------------------
// Fused flash attention over one (b,h): online softmax, scores never hit HBM.
// Q: [bh, l, 64] f16, K: [bh, s, 64] f16, Vt: [bh, 64, s] f16.
// Block = 128 threads (4 waves); each wave owns 16 query rows. K/V tiles for
// each 32-key chunk are staged into double-buffered LDS via async global->LDS
// DMA, overlapped with the previous chunk's WMMAs.
// ---------------------------------------------------------------------------
__global__ void flash_attn(const _Float16* __restrict__ Qh,
                           const _Float16* __restrict__ Kh,
                           const _Float16* __restrict__ Vt,
                           _Float16* __restrict__ Ao) {
  constexpr int L = 2048, S = 2048, E = 64, H = 8, D = 512;
  constexpr int KSTR = 72;   // 64 + 8 halves pad -> conflict-free ds_load_b128
  constexpr int VSTR = 40;   // 32 + 8 halves pad
  constexpr int PCOLS = 40;
  // exp(SCALE*x) == exp2(C2*x), C2 = SCALE * log2(e), SCALE = 64^-0.5
  constexpr float C2 = 0.125f * 1.44269504088896340736f;
  __shared__ _Float16 kbuf[2][32][KSTR] __attribute__((aligned(16)));
  __shared__ _Float16 vbuf[2][64][VSTR] __attribute__((aligned(16)));
  __shared__ _Float16 lds_p[4][16][PCOLS] __attribute__((aligned(16)));

  const int tid    = threadIdx.x;
  const int lane   = tid & 31;
  const int w      = tid >> 5;
  const int lane16 = lane & 15;
  const int hi     = lane >> 4;
  const int bh     = blockIdx.y;
  const int rowbase = blockIdx.x * 64 + w * 16;

  const _Float16* kbase = Kh + (size_t)bh * S * E;
  const _Float16* vbase = Vt + (size_t)bh * E * S;

  // per-thread fixed (row, part) assignment for cooperative tile staging
  const int kr = tid >> 3, kp = (tid & 7) * 8;   // K rows kr, kr+16
  const int vr = tid >> 2, vp = (tid & 3) * 8;   // V rows vr, vr+32

  auto stage_tiles = [&](int s0, int bf) {
    async_copy_b128((unsigned)(size_t)&kbuf[bf][kr][kp],
                    kbase + (size_t)(s0 + kr) * E + kp);
    async_copy_b128((unsigned)(size_t)&kbuf[bf][kr + 16][kp],
                    kbase + (size_t)(s0 + kr + 16) * E + kp);
    async_copy_b128((unsigned)(size_t)&vbuf[bf][vr][vp],
                    vbase + (size_t)vr * S + s0 + vp);
    async_copy_b128((unsigned)(size_t)&vbuf[bf][vr + 32][vp],
                    vbase + (size_t)(vr + 32) * S + s0 + vp);
  };

  // Q fragments stay resident for the whole S loop (16 rows x 64 e).
  const _Float16* qrow = Qh + ((size_t)bh * L + rowbase + lane16) * E;
  const v16h qa0 = a_frag(qrow, hi);       // e = 0..31
  const v16h qa1 = a_frag(qrow + 32, hi);  // e = 32..63

  float m_s[8], l_s[8];
#pragma unroll
  for (int r = 0; r < 8; ++r) { m_s[r] = -3.0e38f; l_s[r] = 0.f; }
  v8f o[4] = {};

  stage_tiles(0, 0);
  wait_async0();
  __syncthreads();

  for (int s0 = 0, it = 0; s0 < S; s0 += 32, ++it) {
    const int cur = it & 1;
    if (s0 + 32 < S) stage_tiles(s0 + 32, cur ^ 1);  // overlap next chunk DMA

    // ---- scores from LDS K tile (fragments loaded just-in-time to keep
    //      peak VGPR liveness low; no spills) ----
    v8f c0 = {}, c1 = {};
    {
      v16h kb = b_frag_lds(&kbuf[cur][lane16][hi * 16]);
      c0 = wmma_f16(qa0, kb, c0);
      kb = b_frag_lds(&kbuf[cur][lane16][32 + hi * 16]);
      c0 = wmma_f16(qa1, kb, c0);
      kb = b_frag_lds(&kbuf[cur][16 + lane16][hi * 16]);
      c1 = wmma_f16(qa0, kb, c1);
      kb = b_frag_lds(&kbuf[cur][16 + lane16][32 + hi * 16]);
      c1 = wmma_f16(qa1, kb, c1);
    }

    // ---- online softmax; DPP16 butterflies for the 16-lane row reductions ----
#pragma unroll
    for (int r = 0; r < 8; ++r) {
      const float v0 = c0[r], v1 = c1[r];
      const float mx = red16_max(fmaxf(v0, v1));
      const float newm  = fmaxf(m_s[r], mx);
      const float alpha = __builtin_amdgcn_exp2f(C2 * (m_s[r] - newm));
      const float p0 = __builtin_amdgcn_exp2f(C2 * (v0 - newm));
      const float p1 = __builtin_amdgcn_exp2f(C2 * (v1 - newm));
      const float rs = red16_sum(p0 + p1);
      l_s[r] = l_s[r] * alpha + rs;
      m_s[r] = newm;
#pragma unroll
      for (int nb = 0; nb < 4; ++nb) o[nb][r] *= alpha;
      lds_p[w][r + hi * 8][lane16]      = (_Float16)p0;
      lds_p[w][r + hi * 8][lane16 + 16] = (_Float16)p1;
    }
    __syncthreads();
    // reload P as an A fragment (16 x 32, K = key chunk)
    const v16h pa = cat8(*(const v8h*)&lds_p[w][lane16][hi * 16],
                         *(const v8h*)&lds_p[w][lane16][hi * 16 + 8]);

    // ---- O += P @ V from LDS V tile (just-in-time fragments) ----
#pragma unroll
    for (int nb = 0; nb < 4; ++nb) {
      v16h vb = b_frag_lds(&vbuf[cur][nb * 16 + lane16][hi * 16]);
      o[nb] = wmma_f16(pa, vb, o[nb]);
    }

    // next chunk's tiles must be resident before anyone reads them, and the
    // current buffers fully consumed before iteration it+1 overwrites them.
    wait_async0();
    __syncthreads();
  }

  // ---- epilogue: normalize (v_rcp_f32) and write f16 attn out [b,l,h*E+e] ----
  const int b_ = bh / H, h_ = bh % H;
#pragma unroll
  for (int r = 0; r < 8; ++r) {
    const float inv = __builtin_amdgcn_rcpf(l_s[r]);
    const int row = rowbase + r + hi * 8;
#pragma unroll
    for (int nb = 0; nb < 4; ++nb) {
      const int e = nb * 16 + lane16;
      Ao[((size_t)b_ * L + row) * D + h_ * E + e] = (_Float16)(o[nb][r] * inv);
    }
  }
}

// ---------------------------------------------------------------------------
extern "C" void kernel_launch(void* const* d_in, const int* in_sizes, int n_in,
                              void* d_out, int out_size, void* d_ws, size_t ws_size,
                              hipStream_t stream) {
  (void)in_sizes; (void)n_in; (void)out_size; (void)ws_size;
  const float* queries = (const float*)d_in[0];
  const float* keys    = (const float*)d_in[1];
  const float* values  = (const float*)d_in[2];
  const float* Wq = (const float*)d_in[3];
  const float* bq = (const float*)d_in[4];
  const float* Wk = (const float*)d_in[5];
  const float* bk = (const float*)d_in[6];
  const float* Wv = (const float*)d_in[7];
  const float* bv = (const float*)d_in[8];
  const float* Wo = (const float*)d_in[9];
  const float* bo = (const float*)d_in[10];

  constexpr int B = 2, L = 2048, D = 512, H = 8, E = 64;
  const size_t headElems = (size_t)B * H * L * E;  // 2M halves (4 MB)

  _Float16* qh = (_Float16*)d_ws;        // [b,h,l,e]
  _Float16* kh = qh + headElems;         // [b,h,s,e]
  _Float16* vt = kh + headElems;         // [b,h,e,s]
  _Float16* ao = vt + headElems;         // [b,l,h*E]

  const dim3 blk(128);
  const dim3 ggrid(B * L / 64, D / 64);  // 64 x 8 tiles

  proj_gemm<0, float><<<ggrid, blk, 0, stream>>>(queries, Wq, bq, qh);
  proj_gemm<0, float><<<ggrid, blk, 0, stream>>>(keys,    Wk, bk, kh);
  proj_gemm<1, float><<<ggrid, blk, 0, stream>>>(values,  Wv, bv, vt);

  flash_attn<<<dim3(L / 64, B * H), blk, 0, stream>>>(qh, kh, vt, ao);

  proj_gemm<2, _Float16><<<ggrid, blk, 0, stream>>>(ao, Wo, bo, (float*)d_out);
}
